// SpatialMambaBlock_35768487641360
// MI455X (gfx1250) — matrix operand order repaired and checked by reading, not compiled, not run on hardware
//
#include <hip/hip_runtime.h>
#include <hip/hip_bf16.h>

typedef __attribute__((ext_vector_type(16))) _Float16 v16h;
typedef __attribute__((ext_vector_type(8)))  _Float16 v8h;
typedef __attribute__((ext_vector_type(8)))  float    v8f;
typedef unsigned int v4u __attribute__((ext_vector_type(4)));
typedef int          v8i __attribute__((ext_vector_type(8)));
typedef int          v4i __attribute__((ext_vector_type(4)));

#define D_MODEL 256
#define D_INNER 512
#define D_STATE 16
#define DT_RANK 16
#define LSEQ    4096
#define BATCH   2
#define NPOS    (BATCH * LSEQ)           // 8192
#define XPROJ_E (DT_RANK + 2 * D_STATE)  // 48

// ---- CDNA5 async / TDM feature detection -----------------------------------
#if __has_builtin(__builtin_amdgcn_global_load_async_to_lds_b128)
#define HAS_ASYNC_LDS 1
#else
#define HAS_ASYNC_LDS 0
#endif
#if __has_builtin(__builtin_amdgcn_tensor_load_to_lds)
#define HAS_TDM 1
#else
#define HAS_TDM 0
#endif
#if __has_builtin(__builtin_amdgcn_s_wait_asynccnt)
#define WAIT_ASYNC() __builtin_amdgcn_s_wait_asynccnt(0)
#else
#define WAIT_ASYNC() asm volatile("s_wait_asynccnt 0" ::: "memory")
#endif
#if __has_builtin(__builtin_amdgcn_s_wait_tensorcnt)
#define WAIT_TENSOR() __builtin_amdgcn_s_wait_tensorcnt(0)
#else
#define WAIT_TENSOR() asm volatile("s_wait_tensorcnt 0" ::: "memory")
#endif

// Generic -> named address-space pointers via integer round-trip, typed as
// v4i* to match the async-to-LDS builtin prototype (vV4i*1 V4i*3 Ii Ii).
// (low 32 bits of a generic LDS flat address are the LDS byte offset)
#define AS_G(p) ((__attribute__((address_space(1))) v4i*)(unsigned long long)(size_t)(p))
#define AS_L(p) ((__attribute__((address_space(3))) v4i*)(unsigned)(size_t)(p))

// ---------------------------------------------------------------------------
// fp32 -> fp16 pack (weights / activations for WMMA)
// ---------------------------------------------------------------------------
__global__ void f32_to_f16_kernel(const float* __restrict__ src,
                                  _Float16* __restrict__ dst, int n) {
    int i = blockIdx.x * blockDim.x + threadIdx.x;
    if (i < n) dst[i] = (_Float16)src[i];
}

// ---------------------------------------------------------------------------
// LayerNorm over channels; x is (B, C, L) so channel reads are strided by L.
// One block (256 threads) per position; writes f16 activations for GEMM1.
// ---------------------------------------------------------------------------
__global__ void layernorm_kernel(const float* __restrict__ x,
                                 const float* __restrict__ ln_w,
                                 const float* __restrict__ ln_b,
                                 _Float16* __restrict__ h16) {
    int pos = blockIdx.x;            // 0..8191
    int b = pos >> 12;
    int l = pos & (LSEQ - 1);
    int c = threadIdx.x;             // 0..255
    float v = x[((size_t)(b * D_MODEL + c) << 12) + l];

    __shared__ float sbuf[D_MODEL];
    sbuf[c] = v;
    __syncthreads();
    for (int s = D_MODEL / 2; s > 0; s >>= 1) {
        if (c < s) sbuf[c] += sbuf[c + s];
        __syncthreads();
    }
    float mu = sbuf[0] * (1.0f / D_MODEL);
    __syncthreads();
    float d = v - mu;
    sbuf[c] = d * d;
    __syncthreads();
    for (int s = D_MODEL / 2; s > 0; s >>= 1) {
        if (c < s) sbuf[c] += sbuf[c + s];
        __syncthreads();
    }
    float var = sbuf[0] * (1.0f / D_MODEL);
    float r = rsqrtf(var + 1e-5f);
    h16[(size_t)pos * D_MODEL + c] = (_Float16)(d * r * ln_w[c] + ln_b[c]);
}

// ---------------------------------------------------------------------------
// WMMA fragment loaders (v_wmma_f32_16x16x32_f16 layouts, ISA 7.12.2)
// ---------------------------------------------------------------------------
__device__ __forceinline__ v16h load_a_frag(const _Float16* __restrict__ A,
                                            int lda, int m0, int k0, int lane) {
    int row = m0 + (lane & 15);
    int kb  = k0 + ((lane >> 4) << 3);
    const _Float16* p = A + (size_t)row * lda + kb;
    v16h f;
    ((v8h*)&f)[0] = *(const v8h*)(p);
    ((v8h*)&f)[1] = *(const v8h*)(p + 16);
    return f;
}

__device__ __forceinline__ v16h load_b_frag(const _Float16* __restrict__ Bt,
                                            int ldb, int n0, int k0, int lane) {
    int row = n0 + (lane & 15);
    int kb  = k0 + ((lane >> 4) << 3);
    const _Float16* p = Bt + (size_t)row * ldb + kb;
    v16h f;
    ((v8h*)&f)[0] = *(const v8h*)(p);
    ((v8h*)&f)[1] = *(const v8h*)(p + 16);
    return f;
}

// ---------------------------------------------------------------------------
// Generic WMMA GEMM: C[M x N] = A[M x K] * Bt[N x K]^T, f32 accumulate.
// ---------------------------------------------------------------------------
template <int NT>
__global__ void wmma_gemm_kernel(const _Float16* __restrict__ A,
                                 const _Float16* __restrict__ Bt,
                                 float* __restrict__ C,
                                 int K, int lda, int ldb, int ldc) {
    int lane = threadIdx.x & 31;
    int wave = threadIdx.x >> 5;
    int m0 = blockIdx.x << 4;
    int n0 = (blockIdx.y * (blockDim.x >> 5) + wave) * (16 * NT);

    v8f acc[NT] = {};
    for (int k0 = 0; k0 < K; k0 += 32) {
        if (k0 + 32 < K) {
            __builtin_prefetch(A + (size_t)(m0 + (lane & 15)) * lda + k0 + 32, 0, 3);
            __builtin_prefetch(Bt + (size_t)(n0 + (lane & 15)) * ldb + k0 + 32, 0, 3);
        }
        v16h a = load_a_frag(A, lda, m0, k0, lane);
#pragma unroll
        for (int t = 0; t < NT; ++t) {
            v16h b = load_b_frag(Bt, ldb, n0 + (t << 4), k0, lane);
            acc[t] = __builtin_amdgcn_wmma_f32_16x16x32_f16(
                false, a, false, b, (short)0, acc[t], false, false);
        }
    }
    int mrow = m0 + ((lane >> 4) << 3);
    int ncol = lane & 15;
#pragma unroll
    for (int t = 0; t < NT; ++t)
#pragma unroll
        for (int r = 0; r < 8; ++r)
            C[(size_t)(mrow + r) * ldc + n0 + (t << 4) + ncol] = acc[t][r];
}

// ---------------------------------------------------------------------------
// Depthwise causal conv1d (D_CONV=4) over L within each batch, + SiLU.
// ---------------------------------------------------------------------------
__global__ void conv_silu_kernel(const float* __restrict__ xz,
                                 const float* __restrict__ conv_w,
                                 const float* __restrict__ conv_b,
                                 float* __restrict__ xc,
                                 _Float16* __restrict__ xc16) {
    int idx = blockIdx.x * blockDim.x + threadIdx.x;
    if (idx >= NPOS * D_INNER) return;
    int d = idx & (D_INNER - 1);
    int pos = idx >> 9;
    int b = pos >> 12;
    int l = pos & (LSEQ - 1);
    float acc = conv_b[d];
#pragma unroll
    for (int j = 0; j < 4; ++j) {
        int ls = l - 3 + j;
        if (ls >= 0)
            acc += conv_w[d * 4 + j] * xz[(size_t)((b << 12) + ls) * 1024 + d];
    }
    float s = acc / (1.0f + __expf(-acc));
    xc[idx] = s;
    xc16[idx] = (_Float16)s;
}

// ---------------------------------------------------------------------------
// dt = softplus(xdbl[:, :16] @ W_dt^T + b_dt)
// ---------------------------------------------------------------------------
__global__ void dt_kernel(const float* __restrict__ xdbl,
                          const float* __restrict__ W_dt,
                          const float* __restrict__ b_dt,
                          float* __restrict__ dt) {
    int idx = blockIdx.x * blockDim.x + threadIdx.x;
    if (idx >= NPOS * D_INNER) return;
    int d = idx & (D_INNER - 1);
    int pos = idx >> 9;
    const float* xr = xdbl + (size_t)pos * XPROJ_E;
    float acc = b_dt[d];
#pragma unroll
    for (int r = 0; r < DT_RANK; ++r) acc += xr[r] * W_dt[d * DT_RANK + r];
    dt[idx] = (acc > 20.0f) ? acc : logf(1.0f + __expf(acc));
}

// ---------------------------------------------------------------------------
// Selective scan. Block = 256 threads = 16 channels x 16 states of one batch.
// Chunked LDS staging: dt tile via the Tensor Data Mover (TENSOR_LOAD_TO_LDS,
// issued by wave 0, waited with s_wait_tensorcnt), xc/B/C tiles via
// GLOBAL_LOAD_ASYNC_TO_LDS_B128 (s_wait_asynccnt). Fallback: plain loads.
// ---------------------------------------------------------------------------
#define TCH 64
__global__ void scan_kernel(const float* __restrict__ dt,
                            const float* __restrict__ xc,
                            const float* __restrict__ xdbl,
                            const float* __restrict__ A_log,
                            float* __restrict__ y) {
    int b  = blockIdx.x >> 5;                 // 32 blocks per batch
    int d0 = (blockIdx.x & 31) * 16;
    int tid = threadIdx.x;
    int dl = tid >> 4;                        // local channel 0..15
    int n  = tid & 15;                        // state 0..15
    int d  = d0 + dl;

    float A = -__expf(A_log[d * D_STATE + n]);
    float h = 0.0f;

    __shared__ float s_dt[TCH][16];
    __shared__ float s_xc[TCH][16];
    __shared__ float s_B[TCH][16];
    __shared__ float s_C[TCH][16];

    for (int t0 = 0; t0 < LSEQ; t0 += TCH) {
        __syncthreads();
        size_t pos0 = (size_t)(b * LSEQ + t0);

#if HAS_TDM
        // TDM: DMA the dt tile (TCH rows x 16 floats, row stride 512 floats)
        // into LDS. D# per ISA ch8: 2D tensor, data_size=4B.
        if (tid < 32) {
            unsigned lds_addr = (unsigned)(size_t)(&s_dt[0][0]);
            unsigned long long ga =
                (unsigned long long)(size_t)(dt + pos0 * D_INNER + d0);
            v4u g0 = { 1u,                       // count=1, user descriptor
                       lds_addr,                 // lds_addr (bytes)
                       (unsigned)ga,             // global_addr[31:0]
                       (unsigned)(ga >> 32) | 0x80000000u };  // addr[56:32] | type=2
            v8i g1 = { 0x20000,                  // data_size=4B (2<<16)
                       16 << 16,                 // tensor_dim0 = 16
                       TCH << 16,                // tensor_dim1 = TCH
                       16 << 16,                 // tile_dim0 = 16
                       TCH,                      // tile_dim1 = TCH
                       D_INNER,                  // tensor_dim0_stride = 512
                       0, 0 };
            v4i gz = { 0, 0, 0, 0 };
#if __clang_major__ >= 23
            v8i gz8 = { 0, 0, 0, 0, 0, 0, 0, 0 };
            __builtin_amdgcn_tensor_load_to_lds(g0, g1, gz, gz, gz8, 0);
#else
            __builtin_amdgcn_tensor_load_to_lds(g0, g1, gz, gz, 0);
#endif
        }
#endif

#if HAS_ASYNC_LDS
        {
            // one 16-byte chunk per thread per array (256 * 16B = full tile)
            int tt = tid >> 2, cc = (tid & 3) << 2;
            size_t pos = pos0 + tt;
            __builtin_amdgcn_global_load_async_to_lds_b128(
                AS_G(xc + pos * D_INNER + d0 + cc), AS_L(&s_xc[tt][cc]), 0, 0);
            __builtin_amdgcn_global_load_async_to_lds_b128(
                AS_G(xdbl + pos * XPROJ_E + DT_RANK + cc), AS_L(&s_B[tt][cc]), 0, 0);
            __builtin_amdgcn_global_load_async_to_lds_b128(
                AS_G(xdbl + pos * XPROJ_E + DT_RANK + D_STATE + cc),
                AS_L(&s_C[tt][cc]), 0, 0);
#if !HAS_TDM
            __builtin_amdgcn_global_load_async_to_lds_b128(
                AS_G(dt + pos * D_INNER + d0 + cc), AS_L(&s_dt[tt][cc]), 0, 0);
#endif
            WAIT_ASYNC();
        }
#else
        for (int i = tid; i < TCH * 16; i += 256) {
            int tt = i >> 4, cc = i & 15;
            size_t pos = pos0 + tt;
            s_xc[tt][cc] = xc[pos * D_INNER + d0 + cc];
            s_B[tt][cc]  = xdbl[pos * XPROJ_E + DT_RANK + cc];
            s_C[tt][cc]  = xdbl[pos * XPROJ_E + DT_RANK + D_STATE + cc];
#if !HAS_TDM
            s_dt[tt][cc] = dt[pos * D_INNER + d0 + cc];
#endif
        }
#endif

#if HAS_TDM
        if (tid < 32) WAIT_TENSOR();
#endif
        __syncthreads();

        for (int tt = 0; tt < TCH; ++tt) {
            float dtv = s_dt[tt][dl];
            float xv  = s_xc[tt][dl];
            float dA  = __expf(dtv * A);
            h = dA * h + dtv * s_B[tt][n] * xv;
            float c = h * s_C[tt][n];
            c += __shfl_xor(c, 1, 32);
            c += __shfl_xor(c, 2, 32);
            c += __shfl_xor(c, 4, 32);
            c += __shfl_xor(c, 8, 32);
            if (n == 0)
                y[(pos0 + tt) * D_INNER + d] = c;
        }
    }
}

// ---------------------------------------------------------------------------
// y_total = (y + xc * D) * silu(z), z = xz[:, 512:1024]; pack f16 for GEMM3.
// ---------------------------------------------------------------------------
__global__ void gate_kernel(const float* __restrict__ y,
                            const float* __restrict__ xc,
                            const float* __restrict__ xz,
                            const float* __restrict__ Dvec,
                            _Float16* __restrict__ yg16) {
    int idx = blockIdx.x * blockDim.x + threadIdx.x;
    if (idx >= NPOS * D_INNER) return;
    int d = idx & (D_INNER - 1);
    int pos = idx >> 9;
    float z = xz[(size_t)pos * 1024 + D_INNER + d];
    float sz = z / (1.0f + __expf(-z));
    yg16[idx] = (_Float16)((y[idx] + xc[idx] * Dvec[d]) * sz);
}

// ---------------------------------------------------------------------------
// out_proj GEMM with fused residual add and (B,L,C) -> (B,C,H,W) store.
// ---------------------------------------------------------------------------
__global__ void out_gemm_kernel(const _Float16* __restrict__ A,
                                const _Float16* __restrict__ Bt,
                                const float* __restrict__ x,
                                float* __restrict__ out) {
    const int K = D_INNER, lda = D_INNER, ldb = D_INNER;
    int lane = threadIdx.x & 31;
    int wave = threadIdx.x >> 5;
    int m0 = blockIdx.x << 4;
    int n0 = wave << 6;  // 4 N-tiles per wave

    v8f acc[4] = {};
    for (int k0 = 0; k0 < K; k0 += 32) {
        if (k0 + 32 < K) {
            __builtin_prefetch(A + (size_t)(m0 + (lane & 15)) * lda + k0 + 32, 0, 3);
            __builtin_prefetch(Bt + (size_t)(n0 + (lane & 15)) * ldb + k0 + 32, 0, 3);
        }
        v16h a = load_a_frag(A, lda, m0, k0, lane);
#pragma unroll
        for (int t = 0; t < 4; ++t) {
            v16h b = load_b_frag(Bt, ldb, n0 + (t << 4), k0, lane);
            acc[t] = __builtin_amdgcn_wmma_f32_16x16x32_f16(
                false, a, false, b, (short)0, acc[t], false, false);
        }
    }
    int mrow = m0 + ((lane >> 4) << 3);
    int ncol = lane & 15;
#pragma unroll
    for (int t = 0; t < 4; ++t)
#pragma unroll
        for (int r = 0; r < 8; ++r) {
            int m = mrow + r;                 // position index
            int c = n0 + (t << 4) + ncol;     // output channel
            int b = m >> 12;
            int l = m & (LSEQ - 1);
            size_t addr = ((size_t)(b * D_MODEL + c) << 12) + l;
            out[addr] = x[addr] + acc[t][r];
        }
}

// ---------------------------------------------------------------------------
// Host-side orchestration
// ---------------------------------------------------------------------------
extern "C" void kernel_launch(void* const* d_in, const int* in_sizes, int n_in,
                              void* d_out, int out_size, void* d_ws, size_t ws_size,
                              hipStream_t stream) {
    const float* x      = (const float*)d_in[0];
    const float* ln_w   = (const float*)d_in[1];
    const float* ln_b   = (const float*)d_in[2];
    const float* W_in   = (const float*)d_in[3];
    const float* conv_w = (const float*)d_in[4];
    const float* conv_b = (const float*)d_in[5];
    const float* W_xp   = (const float*)d_in[6];
    const float* W_dt   = (const float*)d_in[7];
    const float* b_dt   = (const float*)d_in[8];
    const float* A_log  = (const float*)d_in[9];
    const float* Dvec   = (const float*)d_in[10];
    const float* W_out  = (const float*)d_in[11];
    float* out = (float*)d_out;

    char* w = (char*)d_ws;
    _Float16* h16    = (_Float16*)w;  w += (size_t)NPOS * D_MODEL * 2;
    _Float16* Win16  = (_Float16*)w;  w += (size_t)2 * D_INNER * D_MODEL * 2;
    _Float16* Wxp16  = (_Float16*)w;  w += (size_t)XPROJ_E * D_INNER * 2;
    _Float16* Wout16 = (_Float16*)w;  w += (size_t)D_MODEL * D_INNER * 2;
    float* xz   = (float*)w;          w += (size_t)NPOS * 2 * D_INNER * 4;
    float* xc   = (float*)w;          w += (size_t)NPOS * D_INNER * 4;
    _Float16* xc16 = (_Float16*)w;    w += (size_t)NPOS * D_INNER * 2;
    float* xdbl = (float*)w;          w += (size_t)NPOS * XPROJ_E * 4;
    float* dt   = (float*)w;          w += (size_t)NPOS * D_INNER * 4;
    float* y    = (float*)w;          w += (size_t)NPOS * D_INNER * 4;
    _Float16* yg16 = (_Float16*)w;    w += (size_t)NPOS * D_INNER * 2;

    {
        int n1 = 2 * D_INNER * D_MODEL;
        int n2 = XPROJ_E * D_INNER;
        int n3 = D_MODEL * D_INNER;
        f32_to_f16_kernel<<<(n1 + 255) / 256, 256, 0, stream>>>(W_in, Win16, n1);
        f32_to_f16_kernel<<<(n2 + 255) / 256, 256, 0, stream>>>(W_xp, Wxp16, n2);
        f32_to_f16_kernel<<<(n3 + 255) / 256, 256, 0, stream>>>(W_out, Wout16, n3);
    }

    layernorm_kernel<<<NPOS, D_MODEL, 0, stream>>>(x, ln_w, ln_b, h16);

    wmma_gemm_kernel<4><<<dim3(NPOS / 16, 4), 128, 0, stream>>>(
        h16, Win16, xz, D_MODEL, D_MODEL, D_MODEL, 2 * D_INNER);

    conv_silu_kernel<<<(NPOS * D_INNER + 255) / 256, 256, 0, stream>>>(
        xz, conv_w, conv_b, xc, xc16);

    wmma_gemm_kernel<1><<<dim3(NPOS / 16, 1), 96, 0, stream>>>(
        xc16, Wxp16, xdbl, D_INNER, D_INNER, D_INNER, XPROJ_E);

    dt_kernel<<<(NPOS * D_INNER + 255) / 256, 256, 0, stream>>>(
        xdbl, W_dt, b_dt, dt);

    scan_kernel<<<BATCH * (D_INNER / 16), 256, 0, stream>>>(
        dt, xc, xdbl, A_log, y);

    gate_kernel<<<(NPOS * D_INNER + 255) / 256, 256, 0, stream>>>(
        y, xc, xz, Dvec, yg16);

    out_gemm_kernel<<<NPOS / 16, 128, 0, stream>>>(yg16, Wout16, x, out);

    (void)in_sizes; (void)n_in; (void)out_size; (void)ws_size;
}
